// BahdanauAttention_29987461661407
// MI455X (gfx1250) — matrix operand order repaired and checked by reading, not compiled, not run on hardware
//
#include <hip/hip_runtime.h>
#include <hip/hip_bf16.h>

// ---------------------------------------------------------------------------
// Bahdanau attention, fused for MI455X (gfx1250, wave32, WMMA).
//   B=64, S=2048, ENC=DEC=1024
//   scores[b,s] = v . tanh(W_enc @ enc[b,s] + W_dec @ h[b])   (fused, WMMA bf16)
//   attn = softmax(mask ? scores : -1e9)
//   context[b]  = sum_s attn[b,s] * enc[b,s,:]
// ---------------------------------------------------------------------------

typedef __attribute__((ext_vector_type(16))) __bf16 v16bf;
typedef __attribute__((ext_vector_type(8)))  __bf16 v8bf;
typedef __attribute__((ext_vector_type(8)))  float  v8f;
typedef __attribute__((ext_vector_type(4)))  int    v4i;

typedef v4i __attribute__((address_space(1))) gv4i;   // global int4
typedef v4i __attribute__((address_space(3))) lv4i;   // LDS int4

constexpr int   Bb   = 64;
constexpr int   Ss   = 2048;
constexpr int   Ee   = 1024;
constexpr int   Dd   = 1024;
constexpr float NEGV = -1000000000.0f;

constexpr int ROWS_WG = 128;                     // 8 waves x 16 rows
constexpr int ASTR    = 1032;                    // bf16 elems/row, +8 pad (bank spread)
constexpr int BSTR    = 1032;
constexpr size_t LDS_A   = (size_t)ROWS_WG * ASTR * 2;   // 264192 B
constexpr size_t LDS_B   = (size_t)16 * BSTR * 2;        //  33024 B
constexpr size_t LDS_RED = 8 * 256 * 4;                  //   8192 B
constexpr size_t LDS_VDP = 2048 * 4;                     //   8192 B (v + dec_proj row)
constexpr size_t LDS_TOTAL = LDS_A + LDS_B + LDS_RED + LDS_VDP; // 313600 B <= 320KB/WGP

// --- gfx1250 fast tanh (V_TANH_F32 trans op) -------------------------------
__device__ __forceinline__ float fast_tanh(float x) {
#if __has_builtin(__builtin_amdgcn_tanh_f32)
  return __builtin_amdgcn_tanh_f32(x);
#elif __has_builtin(__builtin_amdgcn_tanhf)
  return __builtin_amdgcn_tanhf(x);
#else
  return tanhf(x);
#endif
}

// --- gfx1250 async global->LDS copy (ASYNCcnt path) ------------------------
// Builtin signature (from hipcc diagnostic): param0 = v4i addrspace(1)*
// (printed as "__device__ *"), param1 = v4i addrspace(3)*, then imm offset,
// imm cpol. Generic->AS casts go through integers; a generic LDS pointer
// carries the LDS byte offset in its low 32 bits (flat aperture rule).
#if __has_builtin(__builtin_amdgcn_global_load_async_to_lds_b128)
#define HAS_ASYNC_LDS 1
__device__ __forceinline__ void async_copy_b128(const void* g, void* lds) {
  gv4i* gp = (gv4i*)(unsigned long long)(uintptr_t)g;
  lv4i* lp = (lv4i*)(unsigned int)(uintptr_t)lds;
  __builtin_amdgcn_global_load_async_to_lds_b128(gp, lp, 0, 0);
}
__device__ __forceinline__ void async_wait0() {
#if __has_builtin(__builtin_amdgcn_s_wait_asynccnt)
  __builtin_amdgcn_s_wait_asynccnt(0);
#else
  asm volatile("s_wait_asynccnt 0" ::: "memory");
#endif
}
#else
#define HAS_ASYNC_LDS 0
#endif

// ---------------------------------------------------------------------------
// Kernel 1: W_enc f32 -> bf16 (2MB, stays L2-resident for the whole GEMM)
// ---------------------------------------------------------------------------
__global__ void wcvt_kernel(const float* __restrict__ W, __bf16* __restrict__ Wb) {
  int i = (blockIdx.x * 256 + threadIdx.x) * 4;
  float4 w = *(const float4*)(W + i);
  __bf16 p[4] = {(__bf16)w.x, (__bf16)w.y, (__bf16)w.z, (__bf16)w.w};
  *(unsigned long long*)(Wb + i) = *(const unsigned long long*)p;
}

// ---------------------------------------------------------------------------
// Kernel 2: dec_proj[b,d] = sum_e h[b,e] * W_dec[d,e]   (tiny GEMV, f32)
// ---------------------------------------------------------------------------
__global__ void decproj_kernel(const float* __restrict__ h,
                               const float* __restrict__ Wd,
                               float* __restrict__ out) {
  int gid = blockIdx.x * 256 + threadIdx.x;      // 65536 threads
  int b = gid >> 10, d = gid & 1023;
  const float* hp = h + (size_t)b * Dd;
  const float* wp = Wd + (size_t)d * Dd;
  float acc = 0.f;
  for (int e = 0; e < Dd; e += 4) {
    float4 hv = *(const float4*)(hp + e);
    float4 wv = *(const float4*)(wp + e);
    acc += hv.x * wv.x + hv.y * wv.y + hv.z * wv.z + hv.w * wv.w;
  }
  out[gid] = acc;
}

// ---------------------------------------------------------------------------
// Kernel 3: fused WMMA scores.
//   Workgroup = 256 thr (8 waves), owns 128 consecutive rows r = b*S+s.
//   A tile (rows, bf16) staged once in LDS; W panels (16 cols x 1024 K)
//   streamed through LDS via async-to-LDS. Per wave: D[16x16] accumulated
//   over K=1024 (32 x v_wmma_f32_16x16x32_bf16), then tanh/dot-v epilogue.
//   D-layout: lane(0-15): VGPR j -> (M=j,  N=lane);
//             lane(16-31): VGPR j -> (M=8+j,N=lane-16)  => per-lane N is fixed.
// ---------------------------------------------------------------------------
__global__ __launch_bounds__(256) void scores_kernel(
    const float*  __restrict__ enc,   // [B,S,E]
    const __bf16* __restrict__ Wb,    // [D,E] bf16
    const float*  __restrict__ decp,  // [B,D]
    const float*  __restrict__ vvec,  // [D]
    const int*    __restrict__ mask,  // [B,S]
    float*        __restrict__ attn_raw) {  // [B,S] masked raw scores
  extern __shared__ char smem[];
  __bf16* sA   = (__bf16*)smem;
  __bf16* sB   = (__bf16*)(smem + LDS_A);
  float*  sRed = (float*)(smem + LDS_A + LDS_B);
  float*  sV   = (float*)(smem + LDS_A + LDS_B + LDS_RED);
  float*  sDP  = sV + 1024;

  const int tid  = threadIdx.x;
  const int wave = tid >> 5;
  const int lane = tid & 31;
  const int l    = lane & 15;       // N index / row-group index
  const int hlf  = lane >> 4;       // lane half (K-phase selector)
  const int rowBase = blockIdx.x * ROWS_WG;
  const int b = rowBase / Ss;       // 16 WGs per batch row; never straddles

  // Preload v and this batch's dec_proj row into LDS.
  {
    int e = tid * 4;
    *(float4*)(sV + e)  = *(const float4*)(vvec + e);
    *(float4*)(sDP + e) = *(const float4*)(decp + (size_t)b * Dd + e);
  }

  // Stage A: 128 rows x 1024 f32 -> bf16 in LDS (padded rows).
  for (int i = tid; i < ROWS_WG * 256; i += 256) {
    int lr = i >> 8;
    int kc = (i & 255) * 4;
    float4 x = *(const float4*)(enc + ((size_t)(rowBase + lr)) * Ee + kc);
    __bf16 p[4] = {(__bf16)x.x, (__bf16)x.y, (__bf16)x.z, (__bf16)x.w};
    *(unsigned long long*)(sA + (size_t)lr * ASTR + kc) = *(const unsigned long long*)p;
  }
  __syncthreads();

  float partial[8];
#pragma unroll
  for (int j = 0; j < 8; ++j) partial[j] = 0.f;

  // Fragment base addresses per documented wave32 bf16 layouts:
  //  A(16x32): lanes 0-15 row M=l hold K {kb*32+0..7, +16..23}; lanes 16-31 K {+8..15, +24..31}
  //  B(32x16): lanes 0-15 col N=l hold K {kb*32+0..7, +16..23}; lanes 16-31 K {+8..15, +24..31}
  const __bf16* aRow = sA + (size_t)(wave * 16 + l) * ASTR + hlf * 8;
  const __bf16* bRow = sB + (size_t)l * BSTR + hlf * 8;

  for (int dt = 0; dt < 64; ++dt) {
    // Stage W panel: 16 columns (rows of Wb) x 1024 K, bf16 (32KB).
#if HAS_ASYNC_LDS
    // 2048 x 16B chunks, async DMA into LDS, tracked on ASYNCcnt.
    for (int c = tid; c < 2048; c += 256) {
      int n   = c >> 7;             // panel row (Wb row dt*16+n)
      int kc8 = (c & 127) * 8;      // bf16 element offset within row
      async_copy_b128(Wb + ((size_t)(dt * 16 + n)) * Ee + kc8,
                      sB + (size_t)n * BSTR + kc8);
    }
    async_wait0();
#else
    for (int i = tid; i < 16 * 256; i += 256) {
      int n  = i >> 8;
      int kc = (i & 255) * 4;
      *(unsigned long long*)(sB + (size_t)n * BSTR + kc) =
          *(const unsigned long long*)(Wb + ((size_t)(dt * 16 + n)) * Ee + kc);
    }
#endif
    // Prefetch next panel into cache while we compute (global_prefetch_b8).
    if (dt < 63) {
      const char* nxt = (const char*)(Wb + ((size_t)(dt + 1) * 16) * Ee);
      __builtin_prefetch(nxt + tid * 128, 0, 0);
    }
    __syncthreads();

    v8f acc = {};
#pragma unroll 4
    for (int kb = 0; kb < 32; ++kb) {
      const __bf16* ap = aRow + kb * 32;
      v8bf a0 = *(const v8bf*)ap;
      v8bf a1 = *(const v8bf*)(ap + 16);
      const __bf16* bp = bRow + kb * 32;
      v8bf b0 = *(const v8bf*)bp;
      v8bf b1 = *(const v8bf*)(bp + 16);
      v16bf af, bfm;
#pragma unroll
      for (int i = 0; i < 8; ++i) {
        af[i] = a0[i];  af[i + 8] = a1[i];
        bfm[i] = b0[i]; bfm[i + 8] = b1[i];
      }
      acc = __builtin_amdgcn_wmma_f32_16x16x32_bf16(
          /*neg_a=*/false, af, /*neg_b=*/false, bfm,
          /*c_mod=*/(short)0, acc, /*reuse_a=*/false, /*reuse_b=*/false);
    }

    // Epilogue: per-lane N = dt*16 + l is constant across the 8 M values.
    const int   n   = dt * 16 + l;
    const float dpv = sDP[n];
    const float vv  = sV[n];
#pragma unroll
    for (int j = 0; j < 8; ++j) partial[j] += fast_tanh(acc[j] + dpv) * vv;

    __syncthreads();  // protect sB before next panel overwrite
  }

  // Cross-lane reduction of per-row partial sums (16 N-lanes per row M).
  float* red = sRed + wave * 256;
#pragma unroll
  for (int j = 0; j < 8; ++j) red[(hlf * 8 + j) * 16 + l] = partial[j];
  __syncthreads();
  if (hlf == 0) {
    float s = 0.f;
#pragma unroll
    for (int n2 = 0; n2 < 16; ++n2) s += red[l * 16 + n2];
    int row = rowBase + wave * 16 + l;   // == b*S + s flat index
    attn_raw[row] = (mask[row] == 0) ? NEGV : s;
  }
}

// ---------------------------------------------------------------------------
// Kernel 4: row softmax over S=2048 (one WG per batch row, values in regs)
// ---------------------------------------------------------------------------
__global__ void softmax_kernel(float* __restrict__ attn) {
  __shared__ float sred[256];
  const int b = blockIdx.x, tid = threadIdx.x;
  float* row = attn + (size_t)b * Ss;

  float vals[8];
#pragma unroll
  for (int i = 0; i < 8; ++i) vals[i] = row[tid + i * 256];

  float m = vals[0];
#pragma unroll
  for (int i = 1; i < 8; ++i) m = fmaxf(m, vals[i]);
  sred[tid] = m;
  __syncthreads();
  for (int off = 128; off > 0; off >>= 1) {
    if (tid < off) sred[tid] = fmaxf(sred[tid], sred[tid + off]);
    __syncthreads();
  }
  m = sred[0];
  __syncthreads();

  float sum = 0.f;
#pragma unroll
  for (int i = 0; i < 8; ++i) {
    vals[i] = __expf(vals[i] - m);
    sum += vals[i];
  }
  sred[tid] = sum;
  __syncthreads();
  for (int off = 128; off > 0; off >>= 1) {
    if (tid < off) sred[tid] += sred[tid + off];
    __syncthreads();
  }
  const float inv = 1.f / sred[0];
#pragma unroll
  for (int i = 0; i < 8; ++i) row[tid + i * 256] = vals[i] * inv;
}

// ---------------------------------------------------------------------------
// Kernel 5: context[b,e] = sum_s attn[b,s] * enc[b,s,e]   (HBM-bound pass)
// ---------------------------------------------------------------------------
__global__ void context_kernel(const float* __restrict__ attn,
                               const float* __restrict__ enc,
                               float* __restrict__ ctx) {
  __shared__ float sattn[Ss];
  const int b = blockIdx.x >> 2;                 // 4 blocks per batch
  const int e = ((blockIdx.x & 3) * 256) + threadIdx.x;
  for (int s = threadIdx.x; s < Ss; s += 256) sattn[s] = attn[(size_t)b * Ss + s];
  __syncthreads();
  float acc = 0.f;
  const float* ep = enc + (size_t)b * Ss * Ee + e;
#pragma unroll 4
  for (int s = 0; s < Ss; ++s) acc += sattn[s] * ep[(size_t)s * Ee];
  ctx[(size_t)b * Ee + e] = acc;
}

// ---------------------------------------------------------------------------
// Launch
// ---------------------------------------------------------------------------
extern "C" void kernel_launch(void* const* d_in, const int* in_sizes, int n_in,
                              void* d_out, int out_size, void* d_ws, size_t ws_size,
                              hipStream_t stream) {
  const float* h     = (const float*)d_in[0];   // [B,DEC]
  const float* enc   = (const float*)d_in[1];   // [B,S,ENC]
  const int*   mask  = (const int*)  d_in[2];   // [B,S]
  const float* W_enc = (const float*)d_in[3];   // [DEC,ENC]
  const float* W_dec = (const float*)d_in[4];   // [DEC,DEC]
  const float* vvec  = (const float*)d_in[5];   // [DEC]

  float* ctx  = (float*)d_out;                    // [B,E]
  float* attn = (float*)d_out + (size_t)Bb * Ee;  // [B,S]

  // Workspace: dec_proj (256 KB f32) | W_enc in bf16 (2 MB)
  float*  decp = (float*)d_ws;
  __bf16* Wb   = (__bf16*)((char*)d_ws + (size_t)Bb * Dd * sizeof(float));

  wcvt_kernel<<<(Dd * Ee / 4) / 256, 256, 0, stream>>>(W_enc, Wb);
  decproj_kernel<<<(Bb * Dd) / 256, 256, 0, stream>>>(h, W_dec, decp);
  scores_kernel<<<(Bb * Ss) / ROWS_WG, 256, LDS_TOTAL, stream>>>(
      enc, Wb, decp, vvec, mask, attn);
  softmax_kernel<<<Bb, 256, 0, stream>>>(attn);
  context_kernel<<<Bb * 4, 256, 0, stream>>>(attn, enc, ctx);
}
